// GE2ELoss_51238959841823
// MI455X (gfx1250) — compile-verified, matrix operand order checked by reading, not run on hardware
//
#include <hip/hip_runtime.h>
#include <hip/hip_bf16.h>
#include <math.h>

typedef __attribute__((ext_vector_type(16))) _Float16 v16h;
typedef __attribute__((ext_vector_type(8)))  _Float16 v8h;
typedef __attribute__((ext_vector_type(8)))  float    v8f;

#define N_SPK 1024
#define N_UTT 20
#define D_EMB 512
#define COS_EPS 1e-8f
#define SIM_EPS 1e-6f

// LDS B-tile: 16 columns x 512 K, row stride padded to 520 halves (1040 B)
// so fragment reads hit all 64 LDS banks exactly once per half-wave.
#define BROW 520
#define LDSBUF_BYTES (16 * BROW * 2)   // 16640 B per buffer

// Wait for all outstanding async (global->LDS) transfers of this wave.
#define WAIT_ASYNC0() asm volatile("s_wait_asynccnt 0x0" ::: "memory")

// ---------------------------------------------------------------------------
// Block-wide reduction of three floats (256 threads, wave32), broadcast result.
// ---------------------------------------------------------------------------
__device__ __forceinline__ void block_reduce3(float& a, float& b, float& c,
                                              float* lds) {
  #pragma unroll
  for (int m = 16; m >= 1; m >>= 1) {
    a += __shfl_xor(a, m, 32);
    b += __shfl_xor(b, m, 32);
    c += __shfl_xor(c, m, 32);
  }
  const int lane = threadIdx.x & 31;
  const int wid  = threadIdx.x >> 5;
  if (lane == 0) { lds[wid] = a; lds[8 + wid] = b; lds[16 + wid] = c; }
  __syncthreads();
  if (threadIdx.x == 0) {
    float A = 0.f, B = 0.f, C = 0.f;
    #pragma unroll
    for (int w2 = 0; w2 < 8; w2++) { A += lds[w2]; B += lds[8 + w2]; C += lds[16 + w2]; }
    lds[24] = A; lds[25] = B; lds[26] = C;
  }
  __syncthreads();
  a = lds[24]; b = lds[25]; c = lds[26];
  __syncthreads();
}

// ---------------------------------------------------------------------------
// Phase 1: per-speaker sums, centroid, leave-one-out centroid, norms.
// Writes f16 normalized embeddings e_h [N*M, D], f16 normalized centroids
// c_h [N, D], and f32 diag [N*M] (leave-one-out cosine).
// ---------------------------------------------------------------------------
__global__ __launch_bounds__(256) void ge2e_phase1(
    const float* __restrict__ emb, float* __restrict__ diag,
    _Float16* __restrict__ c_h, _Float16* __restrict__ e_h) {
  __shared__ float lds[32];
  const int n   = blockIdx.x;
  const int tid = threadIdx.x;
  const int d0  = tid * 2;

  const float2* e2 = (const float2*)(emb + (size_t)n * N_UTT * D_EMB);
  float2 ev[N_UTT];
  float s0 = 0.f, s1 = 0.f;
  #pragma unroll
  for (int m = 0; m < N_UTT; m++) {
    float2 v = e2[m * (D_EMB / 2) + tid];
    ev[m] = v; s0 += v.x; s1 += v.y;
  }

  float c0 = s0 * (1.f / N_UTT), c1 = s1 * (1.f / N_UTT);
  float p0 = c0 * c0 + c1 * c1, p1 = 0.f, p2 = 0.f;
  block_reduce3(p0, p1, p2, lds);
  float cinv = 1.f / fmaxf(sqrtf(p0), COS_EPS);
  c_h[(size_t)n * D_EMB + d0]     = (_Float16)(c0 * cinv);
  c_h[(size_t)n * D_EMB + d0 + 1] = (_Float16)(c1 * cinv);

  for (int m = 0; m < N_UTT; m++) {
    float e0 = ev[m].x, e1 = ev[m].y;
    float x0 = (s0 - e0) * (1.f / (N_UTT - 1));
    float x1 = (s1 - e1) * (1.f / (N_UTT - 1));
    float ee = e0 * e0 + e1 * e1;
    float cc = x0 * x0 + x1 * x1;
    float ec = e0 * x0 + e1 * x1;
    block_reduce3(ee, cc, ec, lds);
    float einv = 1.f / fmaxf(sqrtf(ee), COS_EPS);
    float xinv = 1.f / fmaxf(sqrtf(cc), COS_EPS);
    size_t r = (size_t)n * N_UTT + m;
    if (tid == 0) diag[r] = ec * einv * xinv;
    e_h[r * D_EMB + d0]     = (_Float16)(e0 * einv);
    e_h[r * D_EMB + d0 + 1] = (_Float16)(e1 * einv);
  }
}

// ---------------------------------------------------------------------------
// Issue one 16x512 f16 B tile (16 KB) as async global->LDS DMA, no VGPR
// staging. 256 threads x 4 chunks x 16 B (coalesced; padded LDS rows).
// GVS mode: per-lane LDS byte addr in vdst, per-lane 32-bit global offset in
// vaddr, 64-bit base in SGPR pair. Tracked by ASYNCcnt.
// ---------------------------------------------------------------------------
__device__ __forceinline__ void async_tile(unsigned long long cbase,
                                           int ct, const unsigned ldsa[4],
                                           const unsigned cb[4]) {
  #pragma unroll
  for (int j = 0; j < 4; j++) {
    unsigned voff = (unsigned)ct * 16384u + cb[j];
    asm volatile("global_load_async_to_lds_b128 %0, %1, %2"
                 :: "v"(ldsa[j]), "v"(voff), "s"(cbase)
                 : "memory");
  }
}

// ---------------------------------------------------------------------------
// Phase 2: cs = e_n @ c_n^T fused with diagonal replacement + exp-row-sum +
// log-sum-exp. Each wave owns a 16-row strip; A (16x512 f16) resident in 128
// VGPRs. B tiles DMA'd once per workgroup into double-buffered LDS via
// global_load_async_to_lds_b128 and shared by all 8 waves.
// ---------------------------------------------------------------------------
__global__ __launch_bounds__(256) void ge2e_phase2(
    const _Float16* __restrict__ e_h, const _Float16* __restrict__ c_h,
    const float* __restrict__ diag, const float* __restrict__ wp,
    const float* __restrict__ bp, float* __restrict__ out) {
  __shared__ _Float16 bsh[2][16 * BROW];   // 2 x 16.25 KB

  const float w = *wp;
  const float b = *bp;
  const int tid   = threadIdx.x;
  const int lane  = tid & 31;
  const int wid   = tid >> 5;
  const int rbase = (blockIdx.x * 8 + wid) * 16;
  const int g8    = (lane >> 4) << 3;  // 0 for lanes 0-15, 8 for lanes 16-31
  const int koff  = g8;                // K sub-offset within a 32-wide slice

  // Per-thread copy-chunk addresses (constant across iterations).
  // LDS generic pointer -> 32-bit LDS byte address (upper aperture bits drop).
  const unsigned lds_base = (unsigned)(unsigned long long)(&bsh[0][0]);
  unsigned ldsa0[4], ldsa1[4], cb[4];
  #pragma unroll
  for (int j = 0; j < 4; j++) {
    int cidx   = tid + 256 * j;          // 1024 chunks of 16 B per tile
    int row    = cidx >> 6;              // 64 chunks per 512-half row
    int within = (cidx & 63) * 16;       // byte offset within row
    ldsa0[j] = lds_base + (unsigned)(row * (BROW * 2) + within);
    ldsa1[j] = ldsa0[j] + LDSBUF_BYTES;
    cb[j]    = (unsigned)(cidx * 16);    // byte offset within global tile
  }
  const unsigned long long cbase = (unsigned long long)(const void*)c_h;

  // Resident A fragments (ISA 16-bit A layout: lanes<16 hold K {0..7,16..23},
  // lanes>=16 hold K {8..15,24..31} of each 32-wide slice).
  const _Float16* Arow = e_h + (size_t)(rbase + (lane & 15)) * D_EMB;
  v16h afrag[16];
  #pragma unroll
  for (int k = 0; k < 16; k++) {
    v8h lo = *(const v8h*)(Arow + k * 32 + koff);
    v8h hi = *(const v8h*)(Arow + k * 32 + koff + 16);
    v16h a;
    #pragma unroll
    for (int i = 0; i < 8; i++) { a[i] = lo[i]; a[8 + i] = hi[i]; }
    afrag[k] = a;
  }

  // Per-lane row metadata for D-fragment rows (rbase + g8 + i)
  float dg[8]; int sp[8];
  #pragma unroll
  for (int i = 0; i < 8; i++) {
    int r = rbase + g8 + i;
    dg[i] = diag[r];
    sp[i] = r / N_UTT;  // same-speaker column index
  }

  // Prologue: DMA tile 0 into buffer 0, wait, barrier.
  async_tile(cbase, 0, ldsa0, cb);
  WAIT_ASYNC0();
  __syncthreads();

  float rs[8] = {0.f, 0.f, 0.f, 0.f, 0.f, 0.f, 0.f, 0.f};
  const _Float16* Bbase0 = &bsh[0][(lane & 15) * BROW];
  const _Float16* Bbase1 = &bsh[1][(lane & 15) * BROW];

  #pragma unroll 1
  for (int ct = 0; ct < N_SPK / 16; ct++) {
    // Kick off async DMA of the next tile into the other buffer
    // (safe: that buffer was last read in iteration ct-1, barrier passed).
    if (ct + 1 < N_SPK / 16)
      async_tile(cbase, ct + 1, ((ct + 1) & 1) ? ldsa1 : ldsa0, cb);

    const _Float16* Bcol = (ct & 1) ? Bbase1 : Bbase0;
    v8f acc = {};
    #pragma unroll
    for (int k = 0; k < 16; k++) {
      v8h lo = *(const v8h*)(Bcol + k * 32 + koff);
      v8h hi = *(const v8h*)(Bcol + k * 32 + koff + 16);
      v16h bfr;
      #pragma unroll
      for (int i = 0; i < 8; i++) { bfr[i] = lo[i]; bfr[8 + i] = hi[i]; }
      acc = __builtin_amdgcn_wmma_f32_16x16x32_f16(
          /*neg_a=*/false, afrag[k], /*neg_b=*/false, bfr,
          /*c_mod=*/(short)0, acc, /*reuse_a=*/false, /*reuse_b=*/false);
    }

    // Epilogue: lane holds column (ct*16 + (lane&15)), rows g8+0..7 in acc[i].
    const int mycol = ct * 16 + (lane & 15);
    #pragma unroll
    for (int i = 0; i < 8; i++) {
      float v = acc[i];
      if (mycol == sp[i]) v = dg[i];  // leave-one-out cosine on the diagonal
      rs[i] += __expf(w * (v + SIM_EPS) + b);  // sim <= 5 -> exp can't overflow
    }

    // Next tile's DMA (issued above) overlapped with the 16 WMMAs; now drain
    // this wave's async ops and barrier so all waves see the new tile.
    WAIT_ASYNC0();
    __syncthreads();
  }

  // Reduce row sums over the 16 column-lanes, then finish LSE - pos.
  float part = 0.f;
  #pragma unroll
  for (int i = 0; i < 8; i++) {
    float v = rs[i];
    v += __shfl_xor(v, 1, 32);
    v += __shfl_xor(v, 2, 32);
    v += __shfl_xor(v, 4, 32);
    v += __shfl_xor(v, 8, 32);
    if ((lane & 15) == 0) {
      float lse = logf(v + SIM_EPS);
      float pos = w * (dg[i] + SIM_EPS) + b;
      part += lse - pos;
    }
  }
  part += __shfl_xor(part, 16, 32);  // lane 0 (rows 0-7) + lane 16 (rows 8-15)
  if (lane == 0) atomicAdd(out, part);
}

// ---------------------------------------------------------------------------
extern "C" void kernel_launch(void* const* d_in, const int* in_sizes, int n_in,
                              void* d_out, int out_size, void* d_ws, size_t ws_size,
                              hipStream_t stream) {
  const float* emb = (const float*)d_in[0];
  const float* wp  = (const float*)d_in[1];
  const float* bp  = (const float*)d_in[2];
  float* out = (float*)d_out;

  // Workspace layout (256B aligned):
  //   diag : 20480 f32   =    81,920 B
  //   c_h  : 1024*512 f16 = 1,048,576 B
  //   e_h  : 20480*512 f16 = 20,971,520 B   (total ~22.1 MB)
  char* ws = (char*)d_ws;
  float*    diag = (float*)ws;
  _Float16* c_h  = (_Float16*)(ws + 81920);
  _Float16* e_h  = (_Float16*)(ws + 81920 + (size_t)N_SPK * D_EMB * 2);

  hipMemsetAsync(d_out, 0, sizeof(float), stream);
  ge2e_phase1<<<dim3(N_SPK), dim3(256), 0, stream>>>(emb, diag, c_h, e_h);
  ge2e_phase2<<<dim3((N_SPK * N_UTT) / 128), dim3(256), 0, stream>>>(
      e_h, c_h, diag, wp, bp, out);
}
